// Seq2Seq_39685497815703
// MI455X (gfx1250) — matrix operand order repaired and checked by reading, not compiled, not run on hardware
//
#include <hip/hip_runtime.h>
#include <hip/hip_bf16.h>
#include <math.h>

// ---------------- CDNA5 WMMA types ----------------
typedef __attribute__((ext_vector_type(16))) __bf16 v16bf;
typedef __attribute__((ext_vector_type(8)))  __bf16 v8bf;
typedef __attribute__((ext_vector_type(4)))  __bf16 v4bf;
typedef __attribute__((ext_vector_type(8)))  float  v8f;

static __device__ __forceinline__ v16bf cat8(v8bf a, v8bf b) {
    return __builtin_shufflevector(a, b, 0,1,2,3,4,5,6,7,8,9,10,11,12,13,14,15);
}

// CDNA5 async copy: global -> LDS, 16 bytes per lane, tracked by ASYNCcnt.
// (ISA 15.18.3 GLOBAL_LOAD_ASYNC_TO_LDS_B128; generic-pointer low 32 bits are
//  the LDS byte offset per the aperture rules.)
static __device__ __forceinline__ void async_cp16(const __bf16* g, __bf16* l) {
    unsigned lds_off = (unsigned)(unsigned long long)(void*)l;
    asm volatile("global_load_async_to_lds_b128 %0, %1, off"
                 :: "v"(lds_off), "v"((unsigned long long)g)
                 : "memory");
}
static __device__ __forceinline__ void wait_async0() {
    asm volatile("s_wait_asynccnt 0" ::: "memory");
}

// ---------------- constants ----------------
#define Bsz   4096
#define Isz   256
#define Hsz   512
#define Osz   256
#define Ssz   50
#define Tsz   20
#define KTOT  768           // I + H
#define MT    32            // rows per workgroup
#define LSTR  776           // padded LDS stride (bf16 elems): breaks bank alignment
#define LSTR2 520           // padded LDS stride for fc kernel

// ============================================================
// Prep kernels
// ============================================================
__global__ void cvt_bf16_kernel(const float* __restrict__ s, __bf16* __restrict__ d, int n) {
    int i = blockIdx.x * 256 + threadIdx.x;
    if (i < n) d[i] = (__bf16)s[i];
}
__global__ void add_bias_kernel(const float* __restrict__ a, const float* __restrict__ b,
                                float* __restrict__ d, int n) {
    int i = blockIdx.x * 256 + threadIdx.x;
    if (i < n) d[i] = a[i] + b[i];
}
__global__ void zero_kernel(float* __restrict__ p, int n) {
    int i = blockIdx.x * 256 + threadIdx.x;
    if (i < n) p[i] = 0.0f;
}

// ============================================================
// One LSTM timestep: gates = [x_t | h] @ [Wih | Whh]^T + b, then cell update.
//   grid = (B/32, H/128), block = 256 threads (8 waves, wave32)
//   wave w owns hidden cols [hBase + w*16, +16) for ALL FOUR gates
//   -> i/f/g/o accumulators live in the same wave, cell update is in-register.
// A panel staging: x half fp32->bf16 via VALU; h half is bf16 in global and is
// copied with GLOBAL_LOAD_ASYNC_TO_LDS_B128. B fragments stream from L2.
// h is ping-ponged across launches (block reads all of h, writes a slice).
// ============================================================
__global__ __launch_bounds__(256)
void lstm_step_kernel(const float* __restrict__ xBase, int xRowStride, int xColOff,
                      const __bf16* __restrict__ Wih,   // [2048 x 256] row-major (N x K)
                      const __bf16* __restrict__ Whh,   // [2048 x 512] row-major (N x K)
                      const float*  __restrict__ bias,  // [2048] = bih + bhh
                      const __bf16* __restrict__ hIn,   // [B x 512] bf16
                      __bf16* __restrict__ hOut,        // [B x 512] bf16
                      float* __restrict__ cBuf)         // [B x 512] fp32 (in-place, disjoint)
{
    extern __shared__ __bf16 smem[];              // [MT][LSTR]
    const int tid   = threadIdx.x;
    const int mBase = blockIdx.x * MT;
    const int hBase = blockIdx.y * 128;

    const int wave   = tid >> 5;
    const int lane   = tid & 31;
    const int lane15 = lane & 15;
    const int hi     = lane >> 4;          // 0: lanes 0-15, 1: lanes 16-31
    const int nloc   = wave * 16 + lane15; // hidden col within the 128-slice
    const int n      = hBase + nloc;       // global hidden index 0..511

    // B-matrix row pointers for the 4 gates (torch gate order i,f,g,o)
    const __bf16* w1r[4]; const __bf16* w2r[4];
#pragma unroll
    for (int j = 0; j < 4; ++j) {
        int nrow = j * Hsz + n;            // gate column = weight row ([N][K] layout)
        w1r[j] = Wih + (size_t)nrow * Isz;
        w2r[j] = Whh + (size_t)nrow * Hsz;
        __builtin_prefetch(w1r[j], 0, 3);  // global_prefetch_b8: warm first K-tile
        __builtin_prefetch(w2r[j], 0, 3);
    }

    // ---- stage h half of A: pure bf16 copy, async direct-to-LDS ----
    for (int cid = tid; cid < MT * (Hsz / 8); cid += 256) {
        int row = cid >> 6;                // Hsz/8 = 64 chunks per row
        int cc  = (cid & 63) * 8;
        async_cp16(hIn + (size_t)(mBase + row) * Hsz + cc,
                   &smem[row * LSTR + Isz + cc]);
    }
    // ---- stage x half of A: fp32 -> bf16 convert ----
    for (int cid = tid; cid < MT * (Isz / 4); cid += 256) {
        int row = cid >> 6;                // Isz/4 = 64 chunks per row
        int cc  = (cid & 63) * 4;
        float4 v = *(const float4*)(xBase + (size_t)(mBase + row) * xRowStride
                                    + xColOff + cc);
        v4bf b;
        b[0] = (__bf16)v.x; b[1] = (__bf16)v.y; b[2] = (__bf16)v.z; b[3] = (__bf16)v.w;
        *(v4bf*)(&smem[row * LSTR + cc]) = b;
    }
    wait_async0();
    __syncthreads();

    v8f acc[2][4];
#pragma unroll
    for (int i = 0; i < 2; ++i)
#pragma unroll
        for (int j = 0; j < 4; ++j) acc[i][j] = (v8f)(0.0f);

    // A fragment: ISA 16-bit A layout (lanes<16: K{0-7,16-23}; lanes>=16: K{8-15,24-31})
    auto loadA = [&](int mfrag, int kcol) -> v16bf {
        const __bf16* p = &smem[(mfrag * 16 + lane15) * LSTR + kcol + hi * 8];
        return cat8(*(const v8bf*)p, *(const v8bf*)(p + 16));
    };

    // ---- K loop part 1: x @ Wih^T (K = 0..255) ----
#pragma unroll 4
    for (int kt = 0; kt < Isz / 32; ++kt) {
        v16bf a0 = loadA(0, kt * 32);
        v16bf a1 = loadA(1, kt * 32);
#pragma unroll
        for (int j = 0; j < 4; ++j) {
            const __bf16* bp = w1r[j] + kt * 32 + hi * 8;  // K{8h..+7, 16+8h..+7}
            v16bf b = cat8(*(const v8bf*)bp, *(const v8bf*)(bp + 16));
            acc[0][j] = __builtin_amdgcn_wmma_f32_16x16x32_bf16(false, a0, false, b,
                        (short)0, acc[0][j], false, false);
            acc[1][j] = __builtin_amdgcn_wmma_f32_16x16x32_bf16(false, a1, false, b,
                        (short)0, acc[1][j], false, false);
        }
    }
    // ---- K loop part 2: h @ Whh^T (K = 256..767) ----
#pragma unroll 4
    for (int kt = 0; kt < Hsz / 32; ++kt) {
        v16bf a0 = loadA(0, Isz + kt * 32);
        v16bf a1 = loadA(1, Isz + kt * 32);
#pragma unroll
        for (int j = 0; j < 4; ++j) {
            const __bf16* bp = w2r[j] + kt * 32 + hi * 8;
            v16bf b = cat8(*(const v8bf*)bp, *(const v8bf*)(bp + 16));
            acc[0][j] = __builtin_amdgcn_wmma_f32_16x16x32_bf16(false, a0, false, b,
                        (short)0, acc[0][j], false, false);
            acc[1][j] = __builtin_amdgcn_wmma_f32_16x16x32_bf16(false, a1, false, b,
                        (short)0, acc[1][j], false, false);
        }
    }

    // ---- in-register LSTM cell update (i,f,g,o all live in this wave) ----
    const float bi = bias[n];
    const float bf = bias[Hsz + n];
    const float bg = bias[2 * Hsz + n];
    const float bo = bias[3 * Hsz + n];
#pragma unroll
    for (int i = 0; i < 2; ++i) {
#pragma unroll
        for (int r = 0; r < 8; ++r) {
            // C/D layout: VGPR r -> M = r (lanes<16) or 8+r (lanes>=16)
            int grow = mBase + i * 16 + r + hi * 8;
            float iv = acc[i][0][r] + bi;
            float fv = acc[i][1][r] + bf;
            float gv = acc[i][2][r] + bg;
            float ov = acc[i][3][r] + bo;
            float is = 1.0f / (1.0f + __expf(-iv));
            float fs = 1.0f / (1.0f + __expf(-fv));
            float gt = tanhf(gv);
            float os = 1.0f / (1.0f + __expf(-ov));
            size_t idx = (size_t)grow * Hsz + n;
            float cn = fs * cBuf[idx] + is * gt;
            cBuf[idx] = cn;
            hOut[idx] = (__bf16)(os * tanhf(cn));
        }
    }
}

// ============================================================
// FC step: out = h @ fcW^T + fc_b ; writes d_out[:, t, :] and decoder feedback
//   grid = (B/32, 1), block = 256 (8 waves); wave w -> out cols [w*32, +32)
//   Whole A panel is a bf16 copy -> async direct-to-LDS.
// ============================================================
__global__ __launch_bounds__(256)
void fc_step_kernel(const __bf16* __restrict__ hIn,      // [B x 512] bf16
                    const __bf16* __restrict__ fcW,      // [256 x 512] (N x K)
                    const float*  __restrict__ fcB,      // [256]
                    float* __restrict__ out,             // [B x T x 256]
                    float* __restrict__ decIn,           // [B x 256] fp32 feedback
                    int t)
{
    extern __shared__ __bf16 smem[];                     // [MT][LSTR2]
    const int tid   = threadIdx.x;
    const int mBase = blockIdx.x * MT;

    for (int cid = tid; cid < MT * (Hsz / 8); cid += 256) {
        int row = cid >> 6;
        int cc  = (cid & 63) * 8;
        async_cp16(hIn + (size_t)(mBase + row) * Hsz + cc,
                   &smem[row * LSTR2 + cc]);
    }
    wait_async0();
    __syncthreads();

    const int wave = tid >> 5, lane = tid & 31, lane15 = lane & 15, hi = lane >> 4;

    v8f acc[2][2];
#pragma unroll
    for (int i = 0; i < 2; ++i)
#pragma unroll
        for (int j = 0; j < 2; ++j) acc[i][j] = (v8f)(0.0f);

    const __bf16* wr[2];
    int ncol[2];
#pragma unroll
    for (int j = 0; j < 2; ++j) {
        ncol[j] = wave * 32 + j * 16 + lane15;
        wr[j]   = fcW + (size_t)ncol[j] * Hsz;
    }

    auto loadA = [&](int mfrag, int kcol) -> v16bf {
        const __bf16* p = &smem[(mfrag * 16 + lane15) * LSTR2 + kcol + hi * 8];
        return cat8(*(const v8bf*)p, *(const v8bf*)(p + 16));
    };

#pragma unroll 4
    for (int kt = 0; kt < Hsz / 32; ++kt) {
        v16bf a0 = loadA(0, kt * 32);
        v16bf a1 = loadA(1, kt * 32);
#pragma unroll
        for (int j = 0; j < 2; ++j) {
            const __bf16* bp = wr[j] + kt * 32 + hi * 8;
            v16bf b = cat8(*(const v8bf*)bp, *(const v8bf*)(bp + 16));
            acc[0][j] = __builtin_amdgcn_wmma_f32_16x16x32_bf16(false, a0, false, b,
                        (short)0, acc[0][j], false, false);
            acc[1][j] = __builtin_amdgcn_wmma_f32_16x16x32_bf16(false, a1, false, b,
                        (short)0, acc[1][j], false, false);
        }
    }

#pragma unroll
    for (int j = 0; j < 2; ++j) {
        float bn = fcB[ncol[j]];
#pragma unroll
        for (int i = 0; i < 2; ++i) {
#pragma unroll
            for (int r = 0; r < 8; ++r) {
                int grow = mBase + i * 16 + r + hi * 8;
                float v = acc[i][j][r] + bn;
                out[(size_t)grow * (Tsz * Osz) + t * Osz + ncol[j]] = v;
                decIn[(size_t)grow * Osz + ncol[j]] = v;
            }
        }
    }
}

// ============================================================
// Host orchestration
// ============================================================
extern "C" void kernel_launch(void* const* d_in, const int* in_sizes, int n_in,
                              void* d_out, int out_size, void* d_ws, size_t ws_size,
                              hipStream_t stream) {
    (void)in_sizes; (void)n_in; (void)out_size; (void)ws_size;
    const float* x       = (const float*)d_in[0];
    const float* encWihF = (const float*)d_in[1];
    const float* encWhhF = (const float*)d_in[2];
    const float* encBih  = (const float*)d_in[3];
    const float* encBhh  = (const float*)d_in[4];
    const float* decWihF = (const float*)d_in[5];
    const float* decWhhF = (const float*)d_in[6];
    const float* decBih  = (const float*)d_in[7];
    const float* decBhh  = (const float*)d_in[8];
    const float* fcWF    = (const float*)d_in[9];
    const float* fcB     = (const float*)d_in[10];
    float* out = (float*)d_out;

    // workspace carve-out (256B aligned)
    char* ws = (char*)d_ws;
    size_t off = 0;
    auto carve = [&](size_t bytes) -> void* {
        void* p = ws + off;
        off = (off + bytes + 255) & ~(size_t)255;
        return p;
    };
    __bf16* wEncWih = (__bf16*)carve((size_t)4 * Hsz * Isz * 2);
    __bf16* wEncWhh = (__bf16*)carve((size_t)4 * Hsz * Hsz * 2);
    __bf16* wDecWih = (__bf16*)carve((size_t)4 * Hsz * Isz * 2);
    __bf16* wDecWhh = (__bf16*)carve((size_t)4 * Hsz * Hsz * 2);
    __bf16* wFc     = (__bf16*)carve((size_t)Osz * Hsz * 2);
    float*  encB    = (float*)carve((size_t)4 * Hsz * 4);
    float*  decB    = (float*)carve((size_t)4 * Hsz * 4);
    __bf16* hA      = (__bf16*)carve((size_t)Bsz * Hsz * 2);
    __bf16* hB      = (__bf16*)carve((size_t)Bsz * Hsz * 2);
    float*  cBuf    = (float*)carve((size_t)Bsz * Hsz * 4);
    float*  decIn   = (float*)carve((size_t)Bsz * Osz * 4);

    auto blks = [](int n) { return (n + 255) / 256; };
    // weight conversion fp32 -> bf16 (layout already [N][K] = WMMA B layout)
    cvt_bf16_kernel<<<blks(4*Hsz*Isz), 256, 0, stream>>>(encWihF, wEncWih, 4*Hsz*Isz);
    cvt_bf16_kernel<<<blks(4*Hsz*Hsz), 256, 0, stream>>>(encWhhF, wEncWhh, 4*Hsz*Hsz);
    cvt_bf16_kernel<<<blks(4*Hsz*Isz), 256, 0, stream>>>(decWihF, wDecWih, 4*Hsz*Isz);
    cvt_bf16_kernel<<<blks(4*Hsz*Hsz), 256, 0, stream>>>(decWhhF, wDecWhh, 4*Hsz*Hsz);
    cvt_bf16_kernel<<<blks(Osz*Hsz),   256, 0, stream>>>(fcWF,    wFc,     Osz*Hsz);
    add_bias_kernel<<<blks(4*Hsz), 256, 0, stream>>>(encBih, encBhh, encB, 4*Hsz);
    add_bias_kernel<<<blks(4*Hsz), 256, 0, stream>>>(decBih, decBhh, decB, 4*Hsz);
    // bf16 zeros are all-zero bytes -> reuse fp32 zero kernel over half the count
    zero_kernel<<<blks(Bsz*Hsz/2), 256, 0, stream>>>((float*)hA, Bsz*Hsz/2);
    zero_kernel<<<blks(Bsz*Hsz),   256, 0, stream>>>(cBuf,  Bsz*Hsz);
    zero_kernel<<<blks(Bsz*Osz),   256, 0, stream>>>(decIn, Bsz*Osz);

    const dim3 gridL(Bsz / MT, Hsz / 128);
    const size_t smemL = (size_t)MT * LSTR * sizeof(__bf16);
    const size_t smemF = (size_t)MT * LSTR2 * sizeof(__bf16);

    __bf16* hIn = hA; __bf16* hOut = hB;
    // ---- encoder: 50 steps ----
    for (int t = 0; t < Ssz; ++t) {
        lstm_step_kernel<<<gridL, 256, smemL, stream>>>(
            x, Ssz * Isz, t * Isz, wEncWih, wEncWhh, encB, hIn, hOut, cBuf);
        __bf16* tmp = hIn; hIn = hOut; hOut = tmp;
    }
    // ---- decoder: 20 autoregressive steps ----
    for (int t = 0; t < Tsz; ++t) {
        lstm_step_kernel<<<gridL, 256, smemL, stream>>>(
            decIn, Osz, 0, wDecWih, wDecWhh, decB, hIn, hOut, cBuf);
        __bf16* tmp = hIn; hIn = hOut; hOut = tmp;
        fc_step_kernel<<<dim3(Bsz / MT, 1), 256, smemF, stream>>>(
            hIn, wFc, fcB, out, decIn, t);
    }
}
